// vqvae_91336774517250
// MI455X (gfx1250) — matrix operand order repaired and checked by reading, not compiled, not run on hardware
//
#include <hip/hip_runtime.h>

// ---------------------------------------------------------------------------
// CDNA5 (gfx1250) VQ-VAE forward.  bf16 WMMA (v_wmma_f32_16x16x32_bf16) for all
// C=128 convolutions (implicit GEMM, NHWC internal layout) and the VQ distance
// GEMM.  fp32 accumulation throughout.  Conv kernel is templated on all shape
// parameters (powers of two) so index math lowers to shifts/masks and the
// tap/kchunk/ntile loops fully unroll around the WMMAs.
// ---------------------------------------------------------------------------

typedef __attribute__((ext_vector_type(16))) __bf16   v16bf;
typedef __attribute__((ext_vector_type(8)))  float    v8f;
typedef __attribute__((ext_vector_type(4)))  unsigned v4u;

union Frag {
    v4u      q[2];
    unsigned u[8];
    v16bf    v;
};

#define CIN   128
#define KC4   4      // 128 channels / 32-wide K chunks

__device__ __forceinline__ unsigned short f2bf(float f) {
    unsigned x = __float_as_uint(f);
    x += 0x7FFFu + ((x >> 16) & 1u);          // round-to-nearest-even
    return (unsigned short)(x >> 16);
}
__device__ __forceinline__ float bf2f(unsigned short h) {
    return __uint_as_float(((unsigned)h) << 16);
}
// relu on two packed bf16 values (negatives -> +0)
__device__ __forceinline__ unsigned bf16x2_relu(unsigned v) {
    unsigned r = v;
    if (r & 0x80000000u) r &= 0x0000FFFFu;
    if (r & 0x00008000u) r &= 0xFFFF0000u;
    return r;
}

// ---------------------------------------------------------------------------
// e1: 3->128 ch, 4x4 stride2 pad1 conv + ReLU, fp32 NCHW in -> bf16 NHWC out.
// One block per output pixel, one thread per output channel (K=48, VALU).
// ---------------------------------------------------------------------------
__global__ __launch_bounds__(128)
void conv_e1(const float* __restrict__ x, const float* __restrict__ w,
             const float* __restrict__ bias, unsigned short* __restrict__ out) {
    unsigned pix = blockIdx.x;          // 16*128*128 pixels
    int c  = threadIdx.x;               // cout
    int wo = pix & 127;
    int ho = (pix >> 7) & 127;
    int b  = pix >> 14;
    float s = bias[c];
    #pragma unroll
    for (int i = 0; i < 3; ++i)
        #pragma unroll
        for (int kh = 0; kh < 4; ++kh) {
            int hi = ho * 2 - 1 + kh;
            if (hi < 0 || hi >= 256) continue;
            #pragma unroll
            for (int kw = 0; kw < 4; ++kw) {
                int wi = wo * 2 - 1 + kw;
                if (wi < 0 || wi >= 256) continue;
                s += x[((b * 3 + i) * 256 + hi) * 256 + wi] *
                     w[((c * 3 + i) * 4 + kh) * 4 + kw];
            }
        }
    s = s > 0.f ? s : 0.f;
    out[(((b << 7) + ho) * 128 + wo) * 128 + c] = f2bf(s);
}

// ---------------------------------------------------------------------------
// Pack fp32 conv / convT weights into WMMA B-fragment order (bf16):
//   out[(((tap*KC + kc)*NT + nt)*32 + lane)*16 + e]
// B 32x16 fragment: lane = half*16 + n,  K_local = half*8 + e (e<8), 16 + half*8 + e-8 (e>=8)
// conv  : w[cout][cin][kh][kw]          (isT = 0)
// convT : w[cin ][cout][kh][kw]         (isT = 1)
// Also used for the codebook (KH=KW=1, Cout=512, NT=32, isT=0).
// ---------------------------------------------------------------------------
__global__ __launch_bounds__(256)
void pack_wB(const float* __restrict__ w, unsigned short* __restrict__ out,
             int KH, int KW, int CoutReal, int NT, int isT, int total) {
    int i = blockIdx.x * 256 + threadIdx.x;
    if (i >= total) return;
    int e    = i & 15;
    int t    = i >> 4;
    int lane = t & 31;  t >>= 5;
    int nt   = t % NT;  t /= NT;
    int kc   = t & 3;
    int tap  = t >> 2;
    int half = lane >> 4;
    int n    = nt * 16 + (lane & 15);
    int c    = kc * 32 + half * 8 + (e & 7) + ((e >> 3) << 4);
    int kh   = tap / KW, kw = tap % KW;
    float val = 0.f;
    if (n < CoutReal) {
        if (!isT) val = w[((n * CIN + c) * KH + kh) * KW + kw];
        else      val = w[((c * CoutReal + n) * KH + kh) * KW + kw];
    }
    out[i] = f2bf(val);
}

__global__ void cb_norms(const float* __restrict__ cb, float* __restrict__ out) {
    int n = blockIdx.x * 64 + threadIdx.x;
    if (n >= 512) return;
    float s = 0.f;
    for (int c = 0; c < 128; ++c) { float v = cb[n * 128 + c]; s += v * v; }
    out[n] = s;
}

// ---------------------------------------------------------------------------
// Generic implicit-GEMM WMMA conv / convT, fully templated on shape.
// M = 16*HOUT*WOUT rows (16 per wave), N = Cout (NTPW 16-wide tiles per wave),
// K = taps * 128 accumulated in 32-wide bf16 chunks.
// A fragment (16x32 bf16): lane = half*16 + m; VGPR0-3 hold K = half*8 + 0..7,
// VGPR4-7 hold K = 16 + half*8 + 0..7  -> two contiguous 16B channel loads.
// Epilogue: +bias, +residual, ReLU, bf16 NHWC out (+optional fp32 NHWC copy)
// or fp32 NCHW out (for the final 3-channel convT).
// ---------------------------------------------------------------------------
template <int HIN, int WIN, int HOUT, int WOUT, int KH, int KW,
          int STRIDE, int PAD, int IST, int NTILES, int NTPW>
__global__ __launch_bounds__(256)
void conv_wmma_kernel(const unsigned short* __restrict__ in,
                      const unsigned short* __restrict__ wfrag,
                      const float* __restrict__ bias,
                      unsigned short* __restrict__ out_bf,
                      float* __restrict__ out_f32,
                      const unsigned short* __restrict__ resid,
                      float* __restrict__ out_nchw,
                      int CoutReal, int reluIn, int reluOut) {
    constexpr int NGROUPS = NTILES / NTPW;
    constexpr int MTILES  = 16 * HOUT * WOUT / 16;
    constexpr int TOTAL   = MTILES * NGROUPS;

    const int wid  = blockIdx.x * (blockDim.x >> 5) + (threadIdx.x >> 5);
    const int lane = threadIdx.x & 31;
    const int half = lane >> 4;
    const int l15  = lane & 15;
    if (wid >= TOTAL) return;

    const int grp   = wid % NGROUPS;      // NGROUPS is a power of two -> mask
    const int mtile = wid / NGROUPS;
    const int nt0   = grp * NTPW;

    // A-row coordinates for this lane (16 consecutive wo: ho,b uniform in wave)
    const unsigned mA = (unsigned)mtile * 16 + l15;
    const int wo = mA % WOUT;             // power-of-two -> mask/shift
    const int ho = (mA / WOUT) % HOUT;
    const int b  = mA / (WOUT * HOUT);

    v8f acc[NTPW];
    #pragma unroll
    for (int i = 0; i < NTPW; ++i)
        #pragma unroll
        for (int r = 0; r < 8; ++r) acc[i][r] = 0.f;

    #pragma unroll
    for (int tap = 0; tap < KH * KW; ++tap) {
        const int kh = tap / KW, kw = tap % KW;
        int hi, wi;
        bool lv = true;
        if (!IST) {
            hi = ho * STRIDE - PAD + kh;
            wi = wo * STRIDE - PAD + kw;
        } else {
            int nh = ho + PAD - kh;
            int nw = wo + PAD - kw;
            if (nh % STRIDE) continue;        // uniform across wave (ho uniform)
            if (nw % STRIDE) lv = false;
            hi = nh / STRIDE;
            wi = nw / STRIDE;
        }
        if (hi < 0 || hi >= HIN) continue;    // uniform across wave
        if (wi < 0 || wi >= WIN) lv = false;

        const unsigned short* abase = in + (((b * HIN + hi) * WIN + wi) * CIN);

        #pragma unroll
        for (int kc = 0; kc < KC4; ++kc) {
            Frag a;
            if (lv) {
                const int c0 = kc * 32 + half * 8;
                a.q[0] = *(const v4u*)(abase + c0);
                a.q[1] = *(const v4u*)(abase + c0 + 16);
                if (reluIn) {
                    #pragma unroll
                    for (int j = 0; j < 8; ++j) a.u[j] = bf16x2_relu(a.u[j]);
                }
            } else {
                #pragma unroll
                for (int j = 0; j < 8; ++j) a.u[j] = 0u;
            }
            #pragma unroll
            for (int ni = 0; ni < NTPW; ++ni) {
                const int nt = nt0 + ni;
                const unsigned short* bp =
                    wfrag + (((tap * KC4 + kc) * NTILES + nt) * 32 + lane) * 16;
                Frag bf;
                bf.q[0] = *(const v4u*)(bp);
                bf.q[1] = *(const v4u*)(bp + 8);
                acc[ni] = __builtin_amdgcn_wmma_f32_16x16x32_bf16(
                    false, a.v, false, bf.v, (short)0, acc[ni], false, false);
            }
        }
    }

    // Epilogue.  D fragment: lane holds column n = nt*16 + l15;
    // VGPR r holds row m_local = half*8 + r.
    #pragma unroll
    for (int ni = 0; ni < NTPW; ++ni) {
        const int n = (nt0 + ni) * 16 + l15;
        const float bn = (n < CoutReal) ? bias[n] : 0.f;
        #pragma unroll
        for (int r = 0; r < 8; ++r) {
            const unsigned mD = (unsigned)mtile * 16 + half * 8 + r;
            const int wo2 = mD % WOUT;
            const int ho2 = (mD / WOUT) % HOUT;
            const int b2  = mD / (WOUT * HOUT);
            float val = acc[ni][r] + bn;
            if (n < CoutReal) {
                const int idx = ((b2 * HOUT + ho2) * WOUT + wo2) * CoutReal + n;
                if (resid) val += bf2f(resid[idx]);
                if (reluOut) val = val > 0.f ? val : 0.f;
                if (out_nchw) {
                    out_nchw[((b2 * CoutReal + n) * HOUT + ho2) * WOUT + wo2] = val;
                } else {
                    out_bf[idx] = f2bf(val);
                    if (out_f32) out_f32[idx] = val;
                }
            }
        }
    }
}

// ---------------------------------------------------------------------------
// VQ: per position argmin over 512 codes of ||z-c||^2 via WMMA z·cbT
// (d2 = ||c||^2 - 2 z·c ; the z·z term is constant per row).
// Writes z and z_q to d_out in [B,W,H,C] order and the bf16 NHWC decoder input.
// ---------------------------------------------------------------------------
__global__ __launch_bounds__(256)
void vq_kernel(const unsigned short* __restrict__ zbf,
               const float* __restrict__ zf32,
               const unsigned short* __restrict__ cbfrag,
               const float* __restrict__ cnorm,
               const float* __restrict__ cb,
               float* __restrict__ z_out,
               float* __restrict__ zq_out,
               unsigned short* __restrict__ dec_out) {
    const int wid  = blockIdx.x * (blockDim.x >> 5) + (threadIdx.x >> 5);
    const int lane = threadIdx.x & 31;
    const int half = lane >> 4;
    const int l15  = lane & 15;
    constexpr int MTILES = 16 * 64 * 64 / 16;        // 4096
    if (wid >= MTILES) return;

    const unsigned mA = (unsigned)wid * 16 + l15;
    const unsigned short* arow = zbf + mA * 128;
    Frag a[KC4];
    #pragma unroll
    for (int kc = 0; kc < KC4; ++kc) {
        const int c0 = kc * 32 + half * 8;
        a[kc].q[0] = *(const v4u*)(arow + c0);
        a[kc].q[1] = *(const v4u*)(arow + c0 + 16);
    }

    float bestv[8];
    int   besti[8];
    #pragma unroll
    for (int r = 0; r < 8; ++r) { bestv[r] = 3.4e38f; besti[r] = 0; }

    for (int nt = 0; nt < 32; ++nt) {
        v8f acc;
        #pragma unroll
        for (int r = 0; r < 8; ++r) acc[r] = 0.f;
        #pragma unroll
        for (int kc = 0; kc < KC4; ++kc) {
            const unsigned short* bp = cbfrag + (((kc * 32 + nt) * 32 + lane) * 16);
            Frag bf;
            bf.q[0] = *(const v4u*)(bp);
            bf.q[1] = *(const v4u*)(bp + 8);
            acc = __builtin_amdgcn_wmma_f32_16x16x32_bf16(
                false, a[kc].v, false, bf.v, (short)0, acc, false, false);
        }
        const int   n  = nt * 16 + l15;
        const float cn = cnorm[n];
        #pragma unroll
        for (int r = 0; r < 8; ++r) {
            const float d2 = cn - 2.f * acc[r];
            if (d2 < bestv[r]) { bestv[r] = d2; besti[r] = n; }
        }
    }

    // cross-lane argmin within each 16-lane half (D rows live per half)
    #pragma unroll
    for (int r = 0; r < 8; ++r) {
        float bv = bestv[r];
        int   bi = besti[r];
        #pragma unroll
        for (int off = 1; off < 16; off <<= 1) {
            float ov = __shfl_xor(bv, off, 32);
            int   oi = __shfl_xor(bi, off, 32);
            if (ov < bv || (ov == bv && oi < bi)) { bv = ov; bi = oi; }
        }
        bestv[r] = bv; besti[r] = bi;
    }

    // Write outputs: row m = wid*16 + half*8 + r, 16 lanes/half split channels.
    #pragma unroll
    for (int r = 0; r < 8; ++r) {
        const unsigned m = (unsigned)wid * 16 + half * 8 + r;
        const int ww = m & 63;
        const int hh = (m >> 6) & 63;
        const int b  = m >> 12;
        const int zo = (((b << 6) + ww) * 64 + hh) * 128;   // [B,W,H,C]
        const int zi = m * 128;                              // NHWC
        const int di = (((b << 6) + hh) * 64 + ww) * 128;    // NHWC
        const int code = besti[r];
        const float* crow = cb + code * 128;
        for (int c = l15; c < 128; c += 16) {
            const float zq = crow[c];
            z_out[zo + c]  = zf32[zi + c];
            zq_out[zo + c] = zq;
            dec_out[di + c] = f2bf(zq);
        }
    }
}

// ---------------------------------------------------------------------------
// Host side
// ---------------------------------------------------------------------------
template <int HIN, int WIN, int HOUT, int WOUT, int KH, int KW,
          int STRIDE, int PAD, int IST, int NTILES, int NTPW>
static void launch_conv(hipStream_t s,
                        const unsigned short* in, const unsigned short* wfrag,
                        const float* bias, unsigned short* out_bf, float* out_f32,
                        const unsigned short* resid, float* out_nchw,
                        int CoutReal, int reluIn, int reluOut) {
    constexpr long waves  = ((long)16 * HOUT * WOUT / 16) * (NTILES / NTPW);
    constexpr int  blocks = (int)((waves + 7) / 8);
    conv_wmma_kernel<HIN, WIN, HOUT, WOUT, KH, KW, STRIDE, PAD, IST, NTILES, NTPW>
        <<<blocks, 256, 0, s>>>(in, wfrag, bias, out_bf, out_f32, resid, out_nchw,
                                CoutReal, reluIn, reluOut);
}

static void launch_pack(hipStream_t s, const float* w, unsigned short* out,
                        int KH, int KW, int CoutReal, int NT, int isT) {
    int total  = KH * KW * KC4 * NT * 32 * 16;
    int blocks = (total + 255) / 256;
    pack_wB<<<blocks, 256, 0, s>>>(w, out, KH, KW, CoutReal, NT, isT, total);
}

extern "C" void kernel_launch(void* const* d_in, const int* in_sizes, int n_in,
                              void* d_out, int out_size, void* d_ws, size_t ws_size,
                              hipStream_t stream) {
    (void)in_sizes; (void)n_in; (void)out_size; (void)ws_size;

    // ---- inputs (setup_inputs dict order) ----
    const float* x      = (const float*)d_in[0];
    const float* e1_w   = (const float*)d_in[1];
    const float* e1_b   = (const float*)d_in[2];
    const float* e2_w   = (const float*)d_in[3];
    const float* e2_b   = (const float*)d_in[4];
    const float* e3_w   = (const float*)d_in[5];
    const float* e3_b   = (const float*)d_in[6];
    const float* d1_w   = (const float*)d_in[7];
    const float* d1_b   = (const float*)d_in[8];
    const float* dt1_w  = (const float*)d_in[9];
    const float* dt1_b  = (const float*)d_in[10];
    const float* dt2_w  = (const float*)d_in[11];
    const float* dt2_b  = (const float*)d_in[12];
    const float* cbk    = (const float*)d_in[13];
    const float* eb1c1w = (const float*)d_in[14];
    const float* eb1c1b = (const float*)d_in[15];
    const float* eb1c2w = (const float*)d_in[16];
    const float* eb1c2b = (const float*)d_in[17];
    const float* eb2c1w = (const float*)d_in[18];
    const float* eb2c1b = (const float*)d_in[19];
    const float* eb2c2w = (const float*)d_in[20];
    const float* eb2c2b = (const float*)d_in[21];
    const float* db1c1w = (const float*)d_in[22];
    const float* db1c1b = (const float*)d_in[23];
    const float* db1c2w = (const float*)d_in[24];
    const float* db1c2b = (const float*)d_in[25];
    const float* db2c1w = (const float*)d_in[26];
    const float* db2c1b = (const float*)d_in[27];
    const float* db2c2w = (const float*)d_in[28];
    const float* db2c2b = (const float*)d_in[29];

    // ---- outputs ----
    float* z_out  = (float*)d_out;                 // [16,64,64,128]
    float* zq_out = z_out + 8388608;               // [16,64,64,128]
    float* g_out  = z_out + 16777216;              // [16,3,256,256] NCHW

    // ---- workspace layout ----
    char* ws = (char*)d_ws;
    size_t off = 0;
    auto alloc = [&](size_t bytes) { size_t r = off; off = (off + bytes + 255) & ~(size_t)255; return r; };
    const size_t bigAct = (size_t)16 * 128 * 128 * 128 * 2;   // 67 MB bf16
    const size_t smAct  = (size_t)16 * 64 * 64 * 128 * 2;     // 16.8 MB bf16
    unsigned short* BIG0 = (unsigned short*)(ws + alloc(bigAct));
    unsigned short* SM0  = (unsigned short*)(ws + alloc(smAct));
    unsigned short* SM1  = (unsigned short*)(ws + alloc(smAct));
    unsigned short* SM2  = (unsigned short*)(ws + alloc(smAct));
    float*          ZF32 = (float*)(ws + alloc((size_t)16 * 64 * 64 * 128 * 4));
    const size_t w3  = (size_t)9 * KC4 * 8 * 32 * 16 * 2;
    const size_t w1  = (size_t)1 * KC4 * 8 * 32 * 16 * 2;
    const size_t w4  = (size_t)16 * KC4 * 8 * 32 * 16 * 2;
    unsigned short* We2   = (unsigned short*)(ws + alloc(w4));
    unsigned short* We3   = (unsigned short*)(ws + alloc(w3));
    unsigned short* Web1a = (unsigned short*)(ws + alloc(w3));
    unsigned short* Web1b = (unsigned short*)(ws + alloc(w1));
    unsigned short* Web2a = (unsigned short*)(ws + alloc(w3));
    unsigned short* Web2b = (unsigned short*)(ws + alloc(w1));
    unsigned short* Wd1   = (unsigned short*)(ws + alloc(w3));
    unsigned short* Wdb1a = (unsigned short*)(ws + alloc(w3));
    unsigned short* Wdb1b = (unsigned short*)(ws + alloc(w1));
    unsigned short* Wdb2a = (unsigned short*)(ws + alloc(w3));
    unsigned short* Wdb2b = (unsigned short*)(ws + alloc(w1));
    unsigned short* Wdt1  = (unsigned short*)(ws + alloc(w4));
    unsigned short* Wdt2  = (unsigned short*)(ws + alloc((size_t)16 * KC4 * 1 * 32 * 16 * 2));
    unsigned short* Wcb   = (unsigned short*)(ws + alloc((size_t)1 * KC4 * 32 * 32 * 16 * 2));
    float*          CN    = (float*)(ws + alloc(512 * 4));

    // ---- weight packing ----
    launch_pack(stream, e2_w,   We2,   4, 4, 128, 8,  0);
    launch_pack(stream, e3_w,   We3,   3, 3, 128, 8,  0);
    launch_pack(stream, eb1c1w, Web1a, 3, 3, 128, 8,  0);
    launch_pack(stream, eb1c2w, Web1b, 1, 1, 128, 8,  0);
    launch_pack(stream, eb2c1w, Web2a, 3, 3, 128, 8,  0);
    launch_pack(stream, eb2c2w, Web2b, 1, 1, 128, 8,  0);
    launch_pack(stream, d1_w,   Wd1,   3, 3, 128, 8,  0);
    launch_pack(stream, db1c1w, Wdb1a, 3, 3, 128, 8,  0);
    launch_pack(stream, db1c2w, Wdb1b, 1, 1, 128, 8,  0);
    launch_pack(stream, db2c1w, Wdb2a, 3, 3, 128, 8,  0);
    launch_pack(stream, db2c2w, Wdb2b, 1, 1, 128, 8,  0);
    launch_pack(stream, dt1_w,  Wdt1,  4, 4, 128, 8,  1);   // convT [in,out,kh,kw]
    launch_pack(stream, dt2_w,  Wdt2,  4, 4, 3,   1,  1);   // convT, cout=3
    launch_pack(stream, cbk,    Wcb,   1, 1, 512, 32, 0);   // codebook [512,128]
    cb_norms<<<8, 64, 0, stream>>>(cbk, CN);

    // ---- encoder ----
    conv_e1<<<16 * 128 * 128, 128, 0, stream>>>(x, e1_w, e1_b, BIG0);
    // e2: 4x4 s2 p1 + relu   128x128 -> 64x64
    launch_conv<128,128,64,64, 4,4,2,1, 0, 8,2>(stream, BIG0, We2, e2_b,
        SM0, nullptr, nullptr, nullptr, 128, 0, 1);
    // e3: 3x3 s1 p1
    launch_conv<64,64,64,64, 3,3,1,1, 0, 8,2>(stream, SM0, We3, e3_b,
        SM1, nullptr, nullptr, nullptr, 128, 0, 0);
    // eb1: relu -> 3x3 p1 -> relu -> 1x1 p0 (+resid)
    launch_conv<64,64,64,64, 3,3,1,1, 0, 8,2>(stream, SM1, Web1a, eb1c1b,
        SM0, nullptr, nullptr, nullptr, 128, 1, 1);
    launch_conv<64,64,64,64, 1,1,1,0, 0, 8,2>(stream, SM0, Web1b, eb1c2b,
        SM2, nullptr, SM1, nullptr, 128, 0, 0);
    // eb2 (final output also kept in fp32 -> z)
    launch_conv<64,64,64,64, 3,3,1,1, 0, 8,2>(stream, SM2, Web2a, eb2c1b,
        SM0, nullptr, nullptr, nullptr, 128, 1, 1);
    launch_conv<64,64,64,64, 1,1,1,0, 0, 8,2>(stream, SM0, Web2b, eb2c2b,
        SM1, ZF32, SM2, nullptr, 128, 0, 0);

    // ---- VQ (writes z, z_q outputs + decoder input) ----
    vq_kernel<<<512, 256, 0, stream>>>(SM1, ZF32, Wcb, CN, cbk,
                                       z_out, zq_out, SM0);

    // ---- decoder ----
    launch_conv<64,64,64,64, 3,3,1,1, 0, 8,2>(stream, SM0, Wd1, d1_b,
        SM1, nullptr, nullptr, nullptr, 128, 0, 0);
    // db1
    launch_conv<64,64,64,64, 3,3,1,1, 0, 8,2>(stream, SM1, Wdb1a, db1c1b,
        SM2, nullptr, nullptr, nullptr, 128, 1, 1);
    launch_conv<64,64,64,64, 1,1,1,0, 0, 8,2>(stream, SM2, Wdb1b, db1c2b,
        SM0, nullptr, SM1, nullptr, 128, 0, 0);
    // db2
    launch_conv<64,64,64,64, 3,3,1,1, 0, 8,2>(stream, SM0, Wdb2a, db2c1b,
        SM2, nullptr, nullptr, nullptr, 128, 1, 1);
    launch_conv<64,64,64,64, 1,1,1,0, 0, 8,2>(stream, SM2, Wdb2b, db2c2b,
        SM1, nullptr, SM0, nullptr, 128, 0, 0);
    // dt1: convT 4x4 s2 p1 + relu   64x64 -> 128x128
    launch_conv<64,64,128,128, 4,4,2,1, 1, 8,2>(stream, SM1, Wdt1, dt1_b,
        BIG0, nullptr, nullptr, nullptr, 128, 0, 1);
    // dt2: convT 4x4 s2 p1 -> 3ch fp32 NCHW (final output g)
    launch_conv<128,128,256,256, 4,4,2,1, 1, 1,1>(stream, BIG0, Wdt2, dt2_b,
        nullptr, nullptr, nullptr, g_out, 3, 0, 0);
}
